// CrossAttentionWithBias_75960791597217
// MI455X (gfx1250) — matrix operand order repaired and checked by reading, not compiled, not run on hardware
//
#include <hip/hip_runtime.h>
#include <hip/hip_bf16.h>

// ---------------------------------------------------------------------------
// Problem constants (from the reference)
// ---------------------------------------------------------------------------
constexpr int Bc = 2;      // batch
constexpr int Qc = 1024;   // query length
constexpr int KVc = 2048;  // key/value length
constexpr int Ec = 1024;   // embed dim
constexpr int Hc = 16;     // heads
constexpr int Dc = 64;     // head dim
constexpr float SCALEc = 0.125f;  // 1/sqrt(64)

typedef __attribute__((ext_vector_type(16))) __bf16 v16bf;
typedef __attribute__((ext_vector_type(8)))  float  v8f;

union Frag16 {
    unsigned short s[16];
    uint4          u[2];
    v16bf          v;
};

// float -> bf16 round-to-nearest-even
__device__ __forceinline__ unsigned short f2bf(float x) {
    unsigned int u = __float_as_uint(x);
    u += 0x7fffu + ((u >> 16) & 1u);
    return (unsigned short)(u >> 16);
}

// LDS byte offset of a __shared__ generic pointer: LDS aperture addresses
// carry the LDS offset in addr[31:0] (ISA 10.2 aperture mapping).
__device__ __forceinline__ unsigned lds_off(const void* p) {
    return (unsigned)(unsigned long long)p;
}

// Async 16-byte global -> LDS copy (CDNA5, tracked by ASYNCcnt).
__device__ __forceinline__ void async_copy_b128(const void* gsrc, const void* ldst) {
    unsigned long long ga = (unsigned long long)gsrc;
    unsigned la = lds_off(ldst);
    asm volatile("global_load_async_to_lds_b128 %0, %1, off"
                 :: "v"(la), "v"(ga) : "memory");
}
__device__ __forceinline__ void wait_async0() {
    asm volatile("s_wait_asynccnt 0x0" ::: "memory");
}

// A-matrix fragment (16x32 bf16, ISA layout):
//   element j in [0,8)  -> K = kbase + j          (p[0..7])
//   element j in [8,16) -> K = kbase + 16 + (j-8) (p[16..23])
// kbase = (lane>>4)*8 folded into p by the caller.
__device__ __forceinline__ v16bf load_a_frag(const unsigned short* p) {
    Frag16 f;
    f.u[0] = *reinterpret_cast<const uint4*>(p);
    f.u[1] = *reinterpret_cast<const uint4*>(p + 16);
    return f.v;
}

// B-matrix fragment (32x16 bf16, ISA layout): same K order as the A fragment,
// kbase = (lane>>4)*8 folded into p by caller.
__device__ __forceinline__ v16bf load_b_frag(const unsigned short* p) {
    Frag16 f;
    f.u[0] = *reinterpret_cast<const uint4*>(p);
    f.u[1] = *reinterpret_cast<const uint4*>(p + 16);
    return f.v;
}

__device__ __forceinline__ v8f wmma_bf16(v16bf a, v16bf b, v8f c) {
    return __builtin_amdgcn_wmma_f32_16x16x32_bf16(false, a, false, b,
                                                   (short)0, c, false, false);
}

// ---------------------------------------------------------------------------
// fp32 -> bf16 conversion (grid-stride)
// ---------------------------------------------------------------------------
__global__ void cvt_bf16_kernel(const float* __restrict__ in,
                                unsigned short* __restrict__ out, int n) {
    for (int i = blockIdx.x * blockDim.x + threadIdx.x; i < n;
         i += gridDim.x * blockDim.x) {
        out[i] = f2bf(in[i]);
    }
}

// ---------------------------------------------------------------------------
// GEMM: out = A[M,Kd](bf16) @ W[N,Kd]^T (bf16) + bias[N](f32)
// Block = 128 threads = 4 waves; block tile 128(M) x 64(N); wave tile 32x64.
// The 64x32 weight tile is staged in LDS once per block with async copies
// (double buffered) and shared by all 4 waves.
// MODE: 0 = bf16 [M,N]   1 = f32 [M,N]   2 = bf16 transposed V store [b,h,d,k]
// ---------------------------------------------------------------------------
template <int MODE>
__launch_bounds__(128)
__global__ void gemm_bf16_kernel(const unsigned short* __restrict__ A,
                                 const unsigned short* __restrict__ W,
                                 const float* __restrict__ bias,
                                 void* __restrict__ out,
                                 int M, int N, int Kd) {
    __shared__ __align__(16) unsigned short wtile[2][64 * 32];  // [n][k]

    const int lane = threadIdx.x & 31;
    const int wave = threadIdx.x >> 5;
    const int half = lane >> 4;
    const int l15  = lane & 15;
    const int m0 = blockIdx.y * 128 + wave * 32;
    const int n0 = blockIdx.x * 64;
    const int tid = threadIdx.x;

    // stage one 64(n) x 32(k) weight tile into LDS (256 x 16B chunks)
    auto stage_w = [&](int kk, int sel) {
#pragma unroll
        for (int c = 0; c < 2; ++c) {
            const int chunk = tid * 2 + c;       // 0..255
            const int nrow = chunk >> 2;         // 4 chunks per 64B row
            const int koff = (chunk & 3) * 8;    // elements
            async_copy_b128(W + (size_t)(n0 + nrow) * Kd + kk + koff,
                            &wtile[sel][nrow * 32 + koff]);
        }
    };

    v8f acc[2][4];
#pragma unroll
    for (int mi = 0; mi < 2; ++mi)
#pragma unroll
        for (int t = 0; t < 4; ++t) acc[mi][t] = v8f{};

    const unsigned short* arow0 = A + (size_t)(m0 + l15) * Kd + half * 8;
    const unsigned short* arow1 = A + (size_t)(m0 + 16 + l15) * Kd + half * 8;

    stage_w(0, 0);
    for (int kk = 0, it = 0; kk < Kd; kk += 32, ++it) {
        wait_async0();
        __syncthreads();
        if (kk + 32 < Kd) stage_w(kk + 32, (it + 1) & 1);

        __builtin_prefetch(arow0 + kk + 128, 0, 0);
        __builtin_prefetch(arow1 + kk + 128, 0, 0);
        const v16bf a0 = load_a_frag(arow0 + kk);
        const v16bf a1 = load_a_frag(arow1 + kk);

        const unsigned short* wb = &wtile[it & 1][0];
#pragma unroll
        for (int t = 0; t < 4; ++t) {
            const v16bf bf = load_b_frag(wb + (t * 16 + l15) * 32 + half * 8);
            acc[0][t] = wmma_bf16(a0, bf, acc[0][t]);
            acc[1][t] = wmma_bf16(a1, bf, acc[1][t]);
        }
    }

#pragma unroll
    for (int mi = 0; mi < 2; ++mi) {
#pragma unroll
        for (int t = 0; t < 4; ++t) {
#pragma unroll
            for (int r = 0; r < 8; ++r) {
                const int row = m0 + mi * 16 + r + half * 8;
                const int col = n0 + t * 16 + l15;
                const float v = acc[mi][t][r] + bias[col];
                if (MODE == 1) {
                    ((float*)out)[(size_t)row * N + col] = v;
                } else if (MODE == 0) {
                    ((unsigned short*)out)[(size_t)row * N + col] = f2bf(v);
                } else {  // V projection stored transposed: [b, h*D+d, kpos]
                    const int bIdx = row / KVc;
                    const int kpos = row % KVc;
                    ((unsigned short*)out)[((size_t)(bIdx * Ec + col)) * KVc +
                                           kpos] = f2bf(v);
                }
            }
        }
    }
}

// ---------------------------------------------------------------------------
// Flash attention. Block = 4 waves sharing one (b,h); wave owns a 16-row
// q tile. K (row-major [k][d]) and V (pre-transposed [d][k]) chunks are
// staged in LDS via double-buffered async copies and shared by all waves.
// Online softmax; P re-laid-out C->A through per-wave LDS scratch.
// ---------------------------------------------------------------------------
__launch_bounds__(128)
__global__ void attn_kernel(const unsigned short* __restrict__ qp,
                            const unsigned short* __restrict__ kp,
                            const unsigned short* __restrict__ vt,
                            const unsigned char* __restrict__ mask,
                            unsigned short* __restrict__ ctx) {
    __shared__ __align__(16) unsigned short kbuf[2][32 * 64];  // [key][d]
    __shared__ __align__(16) unsigned short vbuf[2][64 * 32];  // [d][k]
    __shared__ __align__(16) unsigned short plds[4][16 * 32];  // per-wave P

    const int lane = threadIdx.x & 31;
    const int wave = threadIdx.x >> 5;
    const int half = lane >> 4;
    const int l15  = lane & 15;
    const int tid  = threadIdx.x;

    const int bb = blockIdx.x;          // [0, B*H*16)
    const int bh = bb >> 4;             // b*16 + h
    const int h  = bh & (Hc - 1);
    const int b  = bh >> 4;
    const int qt = (bb & 15) * 4 + wave;

    // stage 32(key) x 64(d) K tile + 64(d) x 32(k) V tile (512 x 16B chunks)
    auto stage_kv = [&](int k0, int sel) {
#pragma unroll
        for (int c = 0; c < 2; ++c) {
            const int chunk = tid * 2 + c;       // 0..255
            {   // K tile: 8 chunks per 128B row
                const int krow = chunk >> 3;
                const int doff = (chunk & 7) * 8;
                async_copy_b128(
                    kp + (size_t)(b * KVc + k0 + krow) * Ec + h * Dc + doff,
                    &kbuf[sel][krow * 64 + doff]);
            }
            {   // V tile: 4 chunks per 64B row
                const int drow = chunk >> 2;
                const int koff = (chunk & 3) * 8;
                async_copy_b128(
                    vt + (size_t)(b * Ec + h * Dc + drow) * KVc + k0 + koff,
                    &vbuf[sel][drow * 32 + koff]);
            }
        }
    };

    // Q tile fragments: A-layout, K-dim = d (64 = 2 steps of 32)
    const unsigned short* qbase =
        qp + ((size_t)(b * Qc + qt * 16 + l15) * Ec) + h * Dc + half * 8;
    const v16bf a0 = load_a_frag(qbase);
    const v16bf a1 = load_a_frag(qbase + 32);

    float m8[8], l8[8];
#pragma unroll
    for (int r = 0; r < 8; ++r) { m8[r] = -3.0e38f; l8[r] = 0.0f; }
    v8f acc[4] = {v8f{}, v8f{}, v8f{}, v8f{}};

    unsigned short* pl = &plds[wave][0];

    stage_kv(0, 0);
    for (int k0 = 0, it = 0; k0 < KVc; k0 += 32, ++it) {
        wait_async0();
        __syncthreads();
        if (k0 + 32 < KVc) stage_kv(k0 + 32, (it + 1) & 1);
        const int sel = it & 1;

        // ---- S = Q @ K^T for 32 keys (two 16-col subtiles, K-dim 64) ----
        v8f s[2];
#pragma unroll
        for (int st = 0; st < 2; ++st) {
            const unsigned short* kb =
                &kbuf[sel][(st * 16 + l15) * 64 + half * 8];
            const v16bf bb0 = load_b_frag(kb);
            const v16bf bb1 = load_b_frag(kb + 32);
            v8f t = wmma_bf16(a0, bb0, v8f{});
            s[st] = wmma_bf16(a1, bb1, t);
        }

        // key_padding_mask: true -> -inf
        const float mk0 = mask[b * KVc + k0 + l15]      ? -3.0e38f : 0.0f;
        const float mk1 = mask[b * KVc + k0 + 16 + l15] ? -3.0e38f : 0.0f;

        float corr[8];
#pragma unroll
        for (int r = 0; r < 8; ++r) {
            float s0 = s[0][r] * SCALEc + mk0;
            float s1 = s[1][r] * SCALEc + mk1;
            // row max over chunk (a row lives in one 16-lane half)
            float mx = fmaxf(s0, s1);
            mx = fmaxf(mx, __shfl_xor(mx, 1));
            mx = fmaxf(mx, __shfl_xor(mx, 2));
            mx = fmaxf(mx, __shfl_xor(mx, 4));
            mx = fmaxf(mx, __shfl_xor(mx, 8));
            const float mn = fmaxf(m8[r], mx);
            corr[r] = __expf(m8[r] - mn);
            float p0 = (s0 < -1.0e37f) ? 0.0f : __expf(s0 - mn);
            float p1 = (s1 < -1.0e37f) ? 0.0f : __expf(s1 - mn);
            const int row = r + half * 8;
            pl[row * 32 + l15]      = f2bf(p0);
            pl[row * 32 + 16 + l15] = f2bf(p1);
            float ps = p0 + p1;
            ps += __shfl_xor(ps, 1);
            ps += __shfl_xor(ps, 2);
            ps += __shfl_xor(ps, 4);
            ps += __shfl_xor(ps, 8);
            l8[r] = l8[r] * corr[r] + ps;
            m8[r] = mn;
        }
#pragma unroll
        for (int t = 0; t < 4; ++t)
#pragma unroll
            for (int r = 0; r < 8; ++r) acc[t][r] *= corr[r];

        __asm__ volatile("" ::: "memory");  // keep LDS store->load order

        // re-load P in A layout (K-dim = 32 keys of this chunk)
        const v16bf pa = load_a_frag(pl + l15 * 32 + half * 8);

        // ---- acc += P @ V (4 d-subtiles; V already [d][k] in LDS) ----
#pragma unroll
        for (int t = 0; t < 4; ++t) {
            const v16bf vf =
                load_b_frag(&vbuf[sel][(t * 16 + l15) * 32 + half * 8]);
            acc[t] = wmma_bf16(pa, vf, acc[t]);
        }
    }

    // normalize, emit context bf16 in [B*Q, E] (row = b*Q+q, col = h*D+d)
#pragma unroll
    for (int t = 0; t < 4; ++t)
#pragma unroll
        for (int r = 0; r < 8; ++r) {
            const int row = b * Qc + qt * 16 + r + half * 8;
            const int col = h * Dc + t * 16 + l15;
            ctx[(size_t)row * Ec + col] = f2bf(acc[t][r] / l8[r]);
        }
}

// ---------------------------------------------------------------------------
// Host-side orchestration
// ---------------------------------------------------------------------------
extern "C" void kernel_launch(void* const* d_in, const int* in_sizes, int n_in,
                              void* d_out, int out_size, void* d_ws,
                              size_t ws_size, hipStream_t stream) {
    (void)in_sizes; (void)n_in; (void)out_size; (void)ws_size;

    const float* query = (const float*)d_in[0];
    const float* key   = (const float*)d_in[1];
    const float* value = (const float*)d_in[2];
    const unsigned char* mask = (const unsigned char*)d_in[3];
    const float* Wq = (const float*)d_in[4];
    const float* bq = (const float*)d_in[5];
    const float* Wk = (const float*)d_in[6];
    const float* bk = (const float*)d_in[7];
    const float* Wv = (const float*)d_in[8];
    const float* bv = (const float*)d_in[9];
    const float* Wo = (const float*)d_in[10];
    const float* bo = (const float*)d_in[11];

    const int SQ = Bc * Qc * Ec;   // 2M elems
    const int SK = Bc * KVc * Ec;  // 4M elems
    const int SW = Ec * Ec;        // 1M elems

    unsigned short* p = (unsigned short*)d_ws;
    unsigned short* qb  = p; p += SQ;
    unsigned short* kb  = p; p += SK;
    unsigned short* vb  = p; p += SK;
    unsigned short* wqb = p; p += SW;
    unsigned short* wkb = p; p += SW;
    unsigned short* wvb = p; p += SW;
    unsigned short* wob = p; p += SW;
    unsigned short* qp  = p; p += SQ;
    unsigned short* kp  = p; p += SK;
    unsigned short* vt  = p; p += SK;   // V projection, transposed [b,h,d,k]
    unsigned short* ctx = p; p += SQ;

    // 1) fp32 -> bf16 conversions
    auto cvt = [&](const float* src, unsigned short* dst, int n) {
        int blocks = (n + 1023) / 1024;
        if (blocks > 2048) blocks = 2048;
        cvt_bf16_kernel<<<blocks, 256, 0, stream>>>(src, dst, n);
    };
    cvt(query, qb, SQ);
    cvt(key, kb, SK);
    cvt(value, vb, SK);
    cvt(Wq, wqb, SW);
    cvt(Wk, wkb, SW);
    cvt(Wv, wvb, SW);
    cvt(Wo, wob, SW);

    // 2) input projections (Q/K normal bf16; V stored transposed)
    {
        dim3 blk(128);
        dim3 gq(Ec / 64, (Bc * Qc) / 128);   // 16 x 16
        dim3 gk(Ec / 64, (Bc * KVc) / 128);  // 16 x 32
        gemm_bf16_kernel<0><<<gq, blk, 0, stream>>>(qb, wqb, bq, qp,
                                                    Bc * Qc, Ec, Ec);
        gemm_bf16_kernel<0><<<gk, blk, 0, stream>>>(kb, wkb, bk, kp,
                                                    Bc * KVc, Ec, Ec);
        gemm_bf16_kernel<2><<<gk, blk, 0, stream>>>(vb, wvb, bv, vt,
                                                    Bc * KVc, Ec, Ec);
    }

    // 3) flash attention: B*H*16 blocks, 4 waves/block (one q tile each)
    attn_kernel<<<Bc * Hc * 16, 128, 0, stream>>>(qp, kp, vt, mask, ctx);

    // 4) output projection -> fp32 d_out
    {
        dim3 blk(128);
        dim3 go(Ec / 64, (Bc * Qc) / 128);
        gemm_bf16_kernel<1><<<go, blk, 0, stream>>>(ctx, wob, bo, d_out,
                                                    Bc * Qc, Ec, Ec);
    }
}
